// gnn_l2o_optimizer_50912542327356
// MI455X (gfx1250) — matrix operand order repaired and checked by reading, not compiled, not run on hardware
//
#include <hip/hip_runtime.h>
#include <hip/hip_bf16.h>

typedef __attribute__((ext_vector_type(16))) _Float16 v16h;
typedef __attribute__((ext_vector_type(8)))  float    v8f;

#define HH 24      // hidden size
#define G4 96      // 4*H gate width

// -------------------------------------------------------------------------
// Kernel 1: fused LSTM step (WMMA gates) + GCN weight matmul (WMMA xw)
// One wave handles a 16-node tile; block = 256 threads = 8 waves = 128 nodes.
// -------------------------------------------------------------------------
__global__ __launch_bounds__(256)
void lstm_gcnw_kernel(const float* __restrict__ x,
                      const float* __restrict__ h0,
                      const float* __restrict__ c0,
                      const float* __restrict__ w_ih,
                      const float* __restrict__ w_hh,
                      const float* __restrict__ b_ih,
                      const float* __restrict__ b_hh,
                      const float* __restrict__ gcn_w,
                      float* __restrict__ out_h,
                      float* __restrict__ out_c,
                      float* __restrict__ xw,
                      int n)
{
    __shared__ float gates[8][16][97];  // per-wave 16x96 gate tile (+pad)
    __shared__ float hbuf[8][16][25];   // per-wave 16x24 h tile (+pad)

    const int lane = threadIdx.x & 31;
    const int wid  = threadIdx.x >> 5;
    const int base = (blockIdx.x * 8 + wid) * 16;

    const int col = lane & 15;   // N index within 16x16 tile / A row
    const int hi  = lane >> 4;   // lane half selects K sub-range

    // ---- B fragments for w_hh^T (6 tiles of 16 gate columns), f32->f16 ----
    // B(32x16) layout: lane holds column N=lane%16, K = (lane/16)*8 + e (e<8), 16 + (lane/16)*8 + e-8 (e>=8)
    v16h bw[6];
#pragma unroll
    for (int t = 0; t < 6; ++t) {
        const int j  = t * 16 + col;   // gate index 0..95
        const int kb = hi * 8;
#pragma unroll
        for (int e = 0; e < 16; ++e) {
            const int k = kb + e + (e >= 8 ? 8 : 0);
            bw[t][e] = (k < HH) ? (_Float16)w_hh[j * HH + k] : (_Float16)0.0f;
        }
    }
    // ---- B fragments for gcn_w^T (2 tiles of 16 cols; cols 24..31 zero) ----
    v16h bg[2];
#pragma unroll
    for (int t = 0; t < 2; ++t) {
        const int j  = t * 16 + col;
        const int kb = hi * 8;
#pragma unroll
        for (int e = 0; e < 16; ++e) {
            const int k = kb + e + (e >= 8 ? 8 : 0);
            bg[t][e] = (j < HH && k < HH) ? (_Float16)gcn_w[j * HH + k]
                                          : (_Float16)0.0f;
        }
    }

    // ---- A fragment: h_prev tile (16x32, K padded 24->32), f32->f16 ----
    // A layout: lanes 0-15 row=lane (K 0-7 in e0..7, K16-23 in e8..15);
    //           lanes 16-31 row=lane-16 (K 8-15 / K 24-31).
    // Row is 24 floats (96 B, 16-B aligned): use float4 vector loads.
    int rowNode = base + col; if (rowNode >= n) rowNode = n - 1;
    v16h a;
    {
        const float4* hp = (const float4*)(h0 + (size_t)rowNode * HH);
        // elements 0..7: K = hi*8 .. hi*8+7 (always < 24)
        const float4 p0 = hp[hi * 2];
        const float4 p1 = hp[hi * 2 + 1];
        a[0] = (_Float16)p0.x; a[1] = (_Float16)p0.y;
        a[2] = (_Float16)p0.z; a[3] = (_Float16)p0.w;
        a[4] = (_Float16)p1.x; a[5] = (_Float16)p1.y;
        a[6] = (_Float16)p1.z; a[7] = (_Float16)p1.w;
        // elements 8..15: K = 16 + hi*8 .. 23+hi*8 (hi=1 half is zero pad)
        if (hi == 0) {
            const float4 p2 = hp[4];
            const float4 p3 = hp[5];
            a[8]  = (_Float16)p2.x; a[9]  = (_Float16)p2.y;
            a[10] = (_Float16)p2.z; a[11] = (_Float16)p2.w;
            a[12] = (_Float16)p3.x; a[13] = (_Float16)p3.y;
            a[14] = (_Float16)p3.z; a[15] = (_Float16)p3.w;
        } else {
#pragma unroll
            for (int e = 8; e < 16; ++e) a[e] = (_Float16)0.0f;
        }
    }

    // x values for the 8 rows this lane's C/D elements cover (M = v + 8*hi)
    float xv[8];
#pragma unroll
    for (int v = 0; v < 8; ++v) {
        int nd = base + v + hi * 8; if (nd >= n) nd = n - 1;
        xv[v] = x[nd];
    }

    // ---- gates = A*B + (x*w_ih + b_ih + b_hh), 6 WMMAs ----
#pragma unroll
    for (int t = 0; t < 6; ++t) {
        const int j = t * 16 + col;
        const float wih  = w_ih[j];
        const float bias = b_ih[j] + b_hh[j];
        v8f c;
#pragma unroll
        for (int v = 0; v < 8; ++v) c[v] = xv[v] * wih + bias;
        v8f d = __builtin_amdgcn_wmma_f32_16x16x32_f16(
            false, a, false, bw[t], (short)0, c, false, false);
#pragma unroll
        for (int v = 0; v < 8; ++v)
            gates[wid][v + hi * 8][t * 16 + col] = d[v];
    }
    __syncthreads();

    // ---- activations: lane ch handles channel ch of 16 nodes ----
    if (lane < HH) {
        const int ch = lane;
        for (int r = 0; r < 16; ++r) {
            const int nd = base + r;
            const float gi = gates[wid][r][ch];
            const float gf = gates[wid][r][HH + ch];
            const float gg = gates[wid][r][2 * HH + ch];
            const float go = gates[wid][r][3 * HH + ch];
            const float ig = 1.0f / (1.0f + __expf(-gi));
            const float fg = 1.0f / (1.0f + __expf(-gf));
            const float og = 1.0f / (1.0f + __expf(-go));
            const float gt = tanhf(gg);
            const float cp = (nd < n) ? c0[(size_t)nd * HH + ch] : 0.0f;
            const float cn = fg * cp + ig * gt;
            const float hn = og * tanhf(cn);
            hbuf[wid][r][ch] = hn;
            if (nd < n) {
                out_h[(size_t)nd * HH + ch] = hn;
                out_c[(size_t)nd * HH + ch] = cn;
            }
        }
    }
    __syncthreads();

    // ---- xw = h @ gcn_w^T, 2 WMMAs ----
    v16h ah;
#pragma unroll
    for (int e = 0; e < 8; ++e) {
        const int k = hi * 8 + e;
        ah[e] = (k < HH) ? (_Float16)hbuf[wid][col][k] : (_Float16)0.0f;
    }
#pragma unroll
    for (int e = 0; e < 8; ++e) {
        const int k = 16 + hi * 8 + e;
        ah[8 + e] = (k < HH) ? (_Float16)hbuf[wid][col][k] : (_Float16)0.0f;
    }
#pragma unroll
    for (int t = 0; t < 2; ++t) {
        v8f c = {};
        v8f d = __builtin_amdgcn_wmma_f32_16x16x32_f16(
            false, ah, false, bg[t], (short)0, c, false, false);
        const int j = t * 16 + col;
        if (j < HH) {
#pragma unroll
            for (int v = 0; v < 8; ++v) {
                const int nd = base + v + hi * 8;
                if (nd < n) xw[(size_t)nd * HH + j] = d[v];
            }
        }
    }
}

// -------------------------------------------------------------------------
// Kernel 2: zero agg + deg (contiguous region of workspace)
// -------------------------------------------------------------------------
__global__ void zero_kernel(float* __restrict__ p, size_t cnt)
{
    size_t i = (size_t)blockIdx.x * blockDim.x + threadIdx.x;
    const size_t stride = (size_t)gridDim.x * blockDim.x;
    for (; i < cnt; i += stride) p[i] = 0.0f;
}

// -------------------------------------------------------------------------
// Kernel 3: degree count (incoming edges per dst)
// -------------------------------------------------------------------------
__global__ void deg_kernel(const int* __restrict__ ei, float* __restrict__ deg,
                           int ne, int n)
{
    const int e = blockIdx.x * blockDim.x + threadIdx.x;
    if (e >= ne) return;
    const int d = ei[ne + e];
    if ((unsigned)d < (unsigned)n) unsafeAtomicAdd(&deg[d], 1.0f);
}

// -------------------------------------------------------------------------
// Kernel 4: deg -> dinv = rsqrt(deg + 1) (self loop), in place
// -------------------------------------------------------------------------
__global__ void dinv_kernel(float* __restrict__ deg, int n)
{
    const int i = blockIdx.x * blockDim.x + threadIdx.x;
    if (i < n) deg[i] = rsqrtf(deg[i] + 1.0f);
}

// -------------------------------------------------------------------------
// Kernel 5: edge message passing, one thread per (edge, channel):
//   agg[dst][ch] += xw[src][ch] * dinv[src]*dinv[dst]
// 76.8M independent threads, each 1 gather + 1 atomic; consecutive lanes
// cover consecutive channels of one edge -> 96-B contiguous bursts into
// the L2-resident xw/agg tables (19.2 MB each, L2 = 192 MB).
// -------------------------------------------------------------------------
__global__ __launch_bounds__(256)
void edge_kernel(const int* __restrict__ ei,
                 const float* __restrict__ dinv,
                 const float* __restrict__ xw,
                 float* __restrict__ agg, int ne)
{
    const long long t = (long long)blockIdx.x * blockDim.x + threadIdx.x;
    const long long total = (long long)ne * HH;
    if (t >= total) return;
    const int e  = (int)(t / HH);
    const int ch = (int)(t % HH);
    const int s = ei[e];
    const int d = ei[ne + e];
    const float norm = dinv[s] * dinv[d];
    unsafeAtomicAdd(&agg[(size_t)d * HH + ch], xw[(size_t)s * HH + ch] * norm);
}

// -------------------------------------------------------------------------
// Kernel 6: self-loop + bias + Linear(24,1) + elementwise gate
// -------------------------------------------------------------------------
__global__ void final_kernel(const float* __restrict__ x,
                             const float* __restrict__ dinv,
                             const float* __restrict__ xw,
                             const float* __restrict__ agg,
                             const float* __restrict__ gcn_b,
                             const float* __restrict__ lin_w,
                             const float* __restrict__ lin_b,
                             float* __restrict__ out, int n)
{
    const int i = blockIdx.x * blockDim.x + threadIdx.x;
    if (i >= n) return;
    const float di = dinv[i];
    const float di2 = di * di;
    float acc = lin_b[0];
#pragma unroll
    for (int ch = 0; ch < HH; ++ch) {
        const float conv = agg[(size_t)i * HH + ch]
                         + xw[(size_t)i * HH + ch] * di2 + gcn_b[ch];
        acc += conv * lin_w[ch];
    }
    out[i] = x[i] * acc;
}

// -------------------------------------------------------------------------
extern "C" void kernel_launch(void* const* d_in, const int* in_sizes, int n_in,
                              void* d_out, int out_size, void* d_ws, size_t ws_size,
                              hipStream_t stream)
{
    const float* x     = (const float*)d_in[0];
    const float* h0    = (const float*)d_in[1];
    const float* c0    = (const float*)d_in[2];
    const int*   ei    = (const int*)  d_in[3];
    const float* w_ih  = (const float*)d_in[4];
    const float* w_hh  = (const float*)d_in[5];
    const float* b_ih  = (const float*)d_in[6];
    const float* b_hh  = (const float*)d_in[7];
    const float* gcn_w = (const float*)d_in[8];
    const float* gcn_b = (const float*)d_in[9];
    const float* lin_w = (const float*)d_in[10];
    const float* lin_b = (const float*)d_in[11];

    const int n  = in_sizes[0];
    const int ne = in_sizes[3] / 2;

    // d_out layout: [gated out (n)] [h (n*H)] [c (n*H)]
    float* out_gate = (float*)d_out;
    float* out_h    = out_gate + n;
    float* out_c    = out_h + (size_t)n * HH;

    // workspace layout: [xw n*H] [agg n*H] [deg/dinv n]
    float* xw  = (float*)d_ws;
    float* agg = xw  + (size_t)n * HH;
    float* deg = agg + (size_t)n * HH;

    // 1) zero agg + deg (contiguous)
    zero_kernel<<<2048, 256, 0, stream>>>(agg, (size_t)n * HH + (size_t)n);

    // 2) fused LSTM + gcn_w matmul (WMMA)
    lstm_gcnw_kernel<<<(n + 127) / 128, 256, 0, stream>>>(
        x, h0, c0, w_ih, w_hh, b_ih, b_hh, gcn_w, out_h, out_c, xw, n);

    // 3) degree count
    deg_kernel<<<(ne + 255) / 256, 256, 0, stream>>>(ei, deg, ne, n);

    // 4) dinv
    dinv_kernel<<<(n + 255) / 256, 256, 0, stream>>>(deg, n);

    // 5) edge scatter: one thread per (edge, channel)
    const long long ework = (long long)ne * HH;
    const int eblocks = (int)((ework + 255) / 256);
    edge_kernel<<<eblocks, 256, 0, stream>>>(ei, deg, xw, agg, ne);

    // 6) finalize
    final_kernel<<<(n + 255) / 256, 256, 0, stream>>>(
        x, deg, xw, agg, gcn_b, lin_w, lin_b, out_gate, n);
}
